// QuantizeModule2D_50525995270698
// MI455X (gfx1250) — matrix operand order repaired and checked by reading, not compiled, not run on hardware
//
#include <hip/hip_runtime.h>

typedef __attribute__((ext_vector_type(16))) __bf16 v16bf;
typedef __attribute__((ext_vector_type(8)))  float  v8f;
typedef __attribute__((ext_vector_type(4)))  unsigned int u32x4;
typedef __attribute__((ext_vector_type(8)))  int    i32x8;
typedef __attribute__((ext_vector_type(4)))  int    i32x4;

#define NPIX   65536      // B*H*W
#define CCH    256        // channels
#define KCB    1024       // codebook entries
#define QOUT   16777216   // B*C*H*W floats (quant_out)

// ---- ws layout (bytes) ----
#define WS_LOSS   0
#define WS_C2     256                       // 1024 floats  (-0.5*|c_k|^2)
#define WS_IDX    8192                      // 65536 ints
#define WS_CBA    (8192 + 262144)           // 64 chunks x 16KB (hi 8KB | lo 8KB)

__device__ __forceinline__ unsigned short bf16_rne_bits(float f) {
  unsigned u = __float_as_uint(f);
  u += 0x7FFFu + ((u >> 16) & 1u);
  return (unsigned short)(u >> 16);
}
__device__ __forceinline__ float bf16_bits_f32(unsigned short h) {
  return __uint_as_float(((unsigned)h) << 16);
}

#define WMMA_BF16(A, B, C) \
  __builtin_amdgcn_wmma_f32_16x16x32_bf16(false, (A), false, (B), (short)0, (C), false, false)

// ---------------------------------------------------------------------------
// TDM: 1D copy of one 16KB codebook chunk (2048 x 8B) global -> LDS.
// ---------------------------------------------------------------------------
__device__ __forceinline__ void tdm_load_chunk(const void* gsrc, unsigned lds_off) {
  unsigned long long ga = (unsigned long long)gsrc;
  u32x4 g0;
  g0[0] = 1u;                                            // count=1, user mode
  g0[1] = lds_off;                                       // lds_addr (bytes)
  g0[2] = (unsigned)(ga & 0xFFFFFFFFu);                  // global_addr[31:0]
  g0[3] = (unsigned)((ga >> 32) & 0x01FFFFFFu)           // global_addr[56:32]
          | 0x80000000u;                                 // type=2 (image)
  i32x8 g1;
  g1[0] = 0x30000;                                       // data_size=3 (8B)
  g1[1] = (int)(2048u << 16);                            // tensor_dim0 = 2048
  g1[2] = (int)(1u << 16);                               // tensor_dim1 = 1
  g1[3] = (int)(2048u << 16);                            // tile_dim0   = 2048
  g1[4] = 1;                                             // tile_dim1   = 1
  g1[5] = 2048;                                          // tensor_dim0_stride
  g1[6] = 0;
  g1[7] = 0;
  i32x4 z4 = {0, 0, 0, 0};
#if __clang_major__ >= 23
  i32x8 z8 = {0, 0, 0, 0, 0, 0, 0, 0};
  __builtin_amdgcn_tensor_load_to_lds(g0, g1, z4, z4, z8, 0);
#else
  __builtin_amdgcn_tensor_load_to_lds(g0, g1, z4, z4, 0);
#endif
}

// ---------------------------------------------------------------------------
// Kernel 1: prep. 64 blocks (one per 16-code chunk) x 256 threads.
// ---------------------------------------------------------------------------
__global__ __launch_bounds__(256) void vq_prep(const float* __restrict__ cb,
                                               float* c2half,
                                               unsigned short* cbA,
                                               float* lossAcc) {
  const int mc  = blockIdx.x;
  const int tid = threadIdx.x;
  if (mc == 0 && tid == 0) *lossAcc = 0.0f;

  if (tid < 16) {                       // squared norms for 16 codes
    const int m = mc * 16 + tid;
    float s = 0.0f;
    for (int c = 0; c < CCH; ++c) { float v = cb[m * CCH + c]; s += v * v; }
    c2half[m] = -0.5f * s;
  }

  const int kc   = tid >> 5;
  const int lane = tid & 31;
  const int m    = mc * 16 + (lane & 15);
  const int hiL  = (lane >> 4) & 1;
  const int base = mc * 8192 + kc * 512 + lane * 16;   // ushort units
  #pragma unroll
  for (int i = 0; i < 16; ++i) {
    const int v = i >> 1, o = i & 1;
    // A 16x32 bf16 layout: lanes0-15: v0-3 K=0..7, v4-7 K=16..23
    //                      lanes16-31: v0-3 K=8..15, v4-7 K=24..31
    int kl = (v < 4 ? 2 * v : 16 + 2 * (v - 4)) + o + (hiL ? 8 : 0);
    const int k = kc * 32 + kl;
    const float f  = cb[m * CCH + k];
    const unsigned short hb = bf16_rne_bits(f);
    const unsigned short lb = bf16_rne_bits(f - bf16_bits_f32(hb));
    cbA[base + i]        = hb;
    cbA[base + 4096 + i] = lb;
  }
}

// ---------------------------------------------------------------------------
// Kernel 2: argmin. 256 blocks x 256 threads (8 waves, 32 pixels/wave).
// Two B tiles + two independent accumulator chains per wave; A tiles (16KB
// per 16-code chunk) TDM'd into double-buffered LDS; A ds_loads pipelined
// one kc ahead.  Score = S - 0.5*c2 (argmax == argmin distance).
// ---------------------------------------------------------------------------
__global__ __launch_bounds__(256) void vq_argmin(const float* __restrict__ x,
                                                 const float* __restrict__ c2half,
                                                 const unsigned short* __restrict__ cbA,
                                                 int* __restrict__ idx_ws) {
  __shared__ __align__(32) unsigned short ldsA[2][8192];   // 2 x 16KB
  __shared__ __align__(32) float ldsC2[KCB];               // 4KB  (-0.5*c2)

  const int lane = threadIdx.x & 31;
  const int wv   = threadIdx.x >> 5;
  const int n0   = blockIdx.x * 256 + wv * 32;  // 32 pixels per wave
  const int b    = n0 >> 12;                    // HW = 4096 (32 | 4096)
  const int p0   = n0 & 4095;
  const int hiL  = (lane >> 4) & 1;

  // stage -0.5*c2 into LDS (published by the first loop barrier)
  {
    const int t4 = threadIdx.x * 4;
    *(float4*)&ldsC2[t4] = *(const float4*)&c2half[t4];
  }

  // ---- stage B operands: 2 tiles x 16 pixels x 256 channels (bf16 hi/lo) ---
  const float* xp = x + ((size_t)b << 20) + p0 + (lane & 15);
  v16bf Bhi[2][8], Blo[2][8];
  #pragma unroll
  for (int t = 0; t < 2; ++t) {
    #pragma unroll
    for (int kc = 0; kc < 8; ++kc) {
      v16bf h, l;
      #pragma unroll
      for (int i = 0; i < 16; ++i) {
        // B 32x16 bf16 layout: lanes0-15 K=0..7,16..23, lanes16-31 K=8..15,24..31
        const int c = kc * 32 + (hiL ? 8 : 0) + (i < 8 ? i : 8 + i);
        const float f = xp[t * 16 + ((size_t)c << 12)];
        const unsigned short hb = bf16_rne_bits(f);
        h[i] = __builtin_bit_cast(__bf16, hb);
        l[i] = __builtin_bit_cast(__bf16, bf16_rne_bits(f - bf16_bits_f32(hb)));
      }
      Bhi[t][kc] = h; Blo[t][kc] = l;
    }
  }

  const unsigned lds0 = (unsigned)(unsigned long long)(&ldsA[0][0]);
  const unsigned lds1 = (unsigned)(unsigned long long)(&ldsA[1][0]);
  const float* c2p = ldsC2 + (hiL ? 8 : 0);

  if (wv == 0) tdm_load_chunk(cbA, lds0);      // prologue DMA: chunk 0

  float best0 = -3.4e38f, best1 = -3.4e38f;
  int   bm0 = 0, bm1 = 0;

  for (int mc = 0; mc < 64; ++mc) {
    if (wv == 0) __builtin_amdgcn_s_wait_tensorcnt(0);   // chunk mc landed
    __syncthreads();                                     // publish buffer
    if (wv == 0 && mc < 63)                              // prefetch next chunk
      tdm_load_chunk(cbA + (size_t)(mc + 1) * 8192, (mc & 1) ? lds0 : lds1);

    const v16bf* Abuf = (const v16bf*)ldsA[mc & 1];      // hi at 0, lo at +256
    const v8f c2v = *(const v8f*)(c2p + mc * 16);        // init with -0.5*c2
    v8f acc0 = c2v, acc1 = c2v;

    v16bf Ah = Abuf[lane];                               // kc=0, pipelined
    v16bf Al = Abuf[256 + lane];
    #pragma unroll
    for (int kc = 0; kc < 8; ++kc) {
      v16bf AhN = Ah, AlN = Al;
      if (kc < 7) {                                      // prefetch next A
        AhN = Abuf[(kc + 1) * 32 + lane];
        AlN = Abuf[256 + (kc + 1) * 32 + lane];
      }
      acc0 = WMMA_BF16(Ah, Bhi[0][kc], acc0);
      acc1 = WMMA_BF16(Ah, Bhi[1][kc], acc1);
      acc0 = WMMA_BF16(Ah, Blo[0][kc], acc0);
      acc1 = WMMA_BF16(Ah, Blo[1][kc], acc1);
      acc0 = WMMA_BF16(Al, Bhi[0][kc], acc0);
      acc1 = WMMA_BF16(Al, Bhi[1][kc], acc1);
      Ah = AhN; Al = AlN;
    }
    const int mb = mc * 16 + (hiL ? 8 : 0);
    #pragma unroll
    for (int r = 0; r < 8; ++r) {
      const float s0 = acc0[r];
      const bool  t0 = s0 > best0;
      bm0   = t0 ? (mb + r) : bm0;
      best0 = t0 ? s0 : best0;
      const float s1 = acc1[r];
      const bool  t1 = s1 > best1;
      bm1   = t1 ? (mb + r) : bm1;
      best1 = t1 ? s1 : best1;
    }
  }

  // combine the two lane halves (codes 0..7+16k vs 8..15+16k)
  {
    const float ob = __shfl_xor(best0, 16, 32);
    const int   oi = __shfl_xor(bm0, 16, 32);
    if (ob > best0 || (ob == best0 && oi < bm0)) { best0 = ob; bm0 = oi; }
  }
  {
    const float ob = __shfl_xor(best1, 16, 32);
    const int   oi = __shfl_xor(bm1, 16, 32);
    if (ob > best1 || (ob == best1 && oi < bm1)) { best1 = ob; bm1 = oi; }
  }
  if (!hiL) {
    idx_ws[n0 + (lane & 15)]      = bm0;
    idx_ws[n0 + 16 + (lane & 15)] = bm1;
  }
}

// ---------------------------------------------------------------------------
// Kernel 3: gather quant_out + fused MSE. 16384 blocks x 256 threads.
// ---------------------------------------------------------------------------
__global__ __launch_bounds__(256) void vq_quant_loss(const float* __restrict__ x,
                                                     const float* __restrict__ cb,
                                                     const int* __restrict__ idx_ws,
                                                     float* __restrict__ out,
                                                     float* lossAcc) {
  const int t  = blockIdx.x * 256 + threadIdx.x;   // [0, 256*16384)
  const int c  = t >> 14;
  const int q  = t & 16383;
  const int n0 = q * 4;
  const int b  = n0 >> 12;
  const int p  = n0 & 4095;
  const size_t off = ((size_t)(b * CCH + c) << 12) + p;

  const float4 xv = *(const float4*)(x + off);
  const int4   id = *(const int4*)(idx_ws + n0);
  float4 qv;
  qv.x = cb[id.x * CCH + c];
  qv.y = cb[id.y * CCH + c];
  qv.z = cb[id.z * CCH + c];
  qv.w = cb[id.w * CCH + c];
  *(float4*)(out + off) = qv;

  float dx = qv.x - xv.x, dy = qv.y - xv.y, dz = qv.z - xv.z, dw = qv.w - xv.w;
  float s = dx * dx + dy * dy + dz * dz + dw * dw;
  #pragma unroll
  for (int o = 16; o > 0; o >>= 1) s += __shfl_xor(s, o, 32);
  if ((threadIdx.x & 31) == 0) atomicAdd(lossAcc, s);
}

// ---------------------------------------------------------------------------
// Kernel 4: finalize losses + indices-as-float. 256 blocks x 256 threads.
// ---------------------------------------------------------------------------
__global__ __launch_bounds__(256) void vq_finalize(const int* __restrict__ idx_ws,
                                                   const float* __restrict__ lossAcc,
                                                   float* __restrict__ out) {
  const int t = blockIdx.x * 256 + threadIdx.x;
  if (t < NPIX) out[QOUT + 2 + t] = (float)idx_ws[t];
  if (t == 0) {
    const float l = *lossAcc * (1.0f / (float)QOUT);
    out[QOUT + 0] = l;   // codebook_loss
    out[QOUT + 1] = l;   // commitment_loss (numerically identical)
  }
}

extern "C" void kernel_launch(void* const* d_in, const int* in_sizes, int n_in,
                              void* d_out, int out_size, void* d_ws, size_t ws_size,
                              hipStream_t stream) {
  const float* x  = (const float*)d_in[0];
  const float* cb = (const float*)d_in[1];
  float* out = (float*)d_out;
  char*  ws  = (char*)d_ws;

  float*          lossAcc = (float*)(ws + WS_LOSS);
  float*          c2half  = (float*)(ws + WS_C2);
  int*            idx_ws  = (int*)(ws + WS_IDX);
  unsigned short* cbA     = (unsigned short*)(ws + WS_CBA);

  vq_prep<<<64, 256, 0, stream>>>(cb, c2half, cbA, lossAcc);
  vq_argmin<<<256, 256, 0, stream>>>(x, c2half, cbA, idx_ws);
  vq_quant_loss<<<16384, 256, 0, stream>>>(x, cb, idx_ws, out, lossAcc);
  vq_finalize<<<256, 256, 0, stream>>>(idx_ws, lossAcc, out);
}
